// Autoformer_86560770884314
// MI455X (gfx1250) — matrix operand order repaired and checked by reading, not compiled, not run on hardware
//
#include <hip/hip_runtime.h>
#include <hip/hip_bf16.h>
#include <stdint.h>

// Autoformer autocorrelation block for MI455X (gfx1250, wave32).
// - FFT correlation replaced by per-batch Gram matrix circular-diagonal sums
//   so all heavy math is v_wmma_f32_16x16x32_bf16.
// - All GEMM tiles staged LDS-side by the Tensor Data Mover (tensor_load_to_lds)
//   with TDM padding reproducing the conflict-free 72-elem LDS row stride,
//   double-buffered and pipelined on TENSORcnt.

typedef __bf16 bf16;
typedef __attribute__((ext_vector_type(16))) __bf16    v16bf;
typedef __attribute__((ext_vector_type(8)))  __bf16    v8bf;
typedef __attribute__((ext_vector_type(8)))  float     v8f;
typedef __attribute__((ext_vector_type(4)))  float     v4f;
typedef __attribute__((ext_vector_type(4)))  uint32_t  v4u;
typedef __attribute__((ext_vector_type(8)))  uint32_t  v8u;

#define D_MODEL 1024
#define SEQ_L   2048
#define BATCH   8
#define TOPK    7
#define BK      64
#define STR     72   // LDS row stride (bf16): 128B row + 16B TDM pad -> conflict-free b128

__device__ __forceinline__ v8f wmma_bf16(v16bf a, v16bf b, v8f c) {
    return __builtin_amdgcn_wmma_f32_16x16x32_bf16(false, a, false, b, (short)0, c, false, false);
}

__device__ __forceinline__ uint32_t lds_addr_of(const void* p) {
    // generic pointer to __shared__: low 32 bits are the LDS byte address
    return (uint32_t)(uintptr_t)p;
}

// ---------------------------------------------------------------------------
// TDM: DMA one 128-row x 64-col bf16 tile (row stride D_MODEL elems in memory)
// into LDS with a 16B pad after every 128B row => LDS row stride = 72 elems.
// D# per CDNA5 ISA 8.3/8.4. Issued by one wave; tracked with TENSORcnt.
// ---------------------------------------------------------------------------
__device__ __forceinline__ void tdm_load_tile(const bf16* g, uint32_t lds_b, uint32_t nrows) {
    const uint64_t ga = (uint64_t)(uintptr_t)g;
    v4u g0;
    g0[0] = 1u;                                         // count=1, user mode
    g0[1] = lds_b;                                      // lds_addr
    g0[2] = (uint32_t)ga;                               // global_addr[31:0]
    g0[3] = (uint32_t)((ga >> 32) & 0x01FFFFFFu)        // global_addr[56:32]
          | (2u << 30);                                 // type=2 ("image")
    v8u g1;
    g1[0] = (1u << 16)                                  // data_size = 2 bytes
          | (1u << 20)                                  // pad_enable
          | (4u << 22)                                  // pad_interval: every 32 DWORDs (128B)
          | (3u << 25);                                 // pad_amount: 4 DWORDs (16B)
    g1[1] = (uint32_t)D_MODEL << 16;                    // tensor_dim0[15:0] @ bits 63:48
    g1[2] = (nrows & 0xFFFFu) << 16;                    // tensor_dim0 hi=0 | tensor_dim1[15:0]
    g1[3] = ((nrows >> 16) & 0xFFFFu) | (64u << 16);    // tensor_dim1 hi | tile_dim0=64
    g1[4] = 128u;                                       // tile_dim1=128 | tile_dim2=0
    g1[5] = (uint32_t)D_MODEL;                          // tensor_dim0_stride[31:0]
    g1[6] = 0u;                                         // stride hi | tensor_dim1_stride lo
    g1[7] = 0u;
    asm volatile("tensor_load_to_lds %0, %1" :: "s"(g0), "s"(g1) : "memory");
}

// A-fragment (16x32 bf16): lane<16 M=lane, K=0..7 & 16..23; lane>=16 K=8..15 & 24..31
__device__ __forceinline__ v16bf frag_a(const bf16* s, int row_base, int k0, int lane) {
    const bf16* p = s + (row_base + (lane & 15)) * STR + k0 + ((lane >> 4) << 3);
    v8bf lo = *(const v8bf*)(p);
    v8bf hi = *(const v8bf*)(p + 16);
    return __builtin_shufflevector(lo, hi, 0,1,2,3,4,5,6,7,8,9,10,11,12,13,14,15);
}
// B-fragment (32x16 bf16): lane%16 = N column, lane<16 K=0..7 & 16..23; lane>=16 K=8..15 & 24..31
__device__ __forceinline__ v16bf frag_b(const bf16* s, int col_base, int k0, int lane) {
    const bf16* p = s + (col_base + (lane & 15)) * STR + k0 + ((lane >> 4) << 3);
    v8bf lo = *(const v8bf*)(p);
    v8bf hi = *(const v8bf*)(p + 16);
    return __builtin_shufflevector(lo, hi, 0,1,2,3,4,5,6,7,8,9,10,11,12,13,14,15);
}

// 32x64 wave tile x one 64-deep K-tile: 16 WMMAs
__device__ __forceinline__ void mma_tile(const bf16* As, const bf16* Bs,
                                         int wm0, int wn0, int lane, v8f acc[2][4]) {
    #pragma unroll
    for (int kk = 0; kk < BK; kk += 32) {
        v16bf a0 = frag_a(As, wm0,      kk, lane);
        v16bf a1 = frag_a(As, wm0 + 16, kk, lane);
        v16bf b0 = frag_b(Bs, wn0,      kk, lane);
        v16bf b1 = frag_b(Bs, wn0 + 16, kk, lane);
        v16bf b2 = frag_b(Bs, wn0 + 32, kk, lane);
        v16bf b3 = frag_b(Bs, wn0 + 48, kk, lane);
        acc[0][0] = wmma_bf16(a0, b0, acc[0][0]);
        acc[0][1] = wmma_bf16(a0, b1, acc[0][1]);
        acc[0][2] = wmma_bf16(a0, b2, acc[0][2]);
        acc[0][3] = wmma_bf16(a0, b3, acc[0][3]);
        acc[1][0] = wmma_bf16(a1, b0, acc[1][0]);
        acc[1][1] = wmma_bf16(a1, b1, acc[1][1]);
        acc[1][2] = wmma_bf16(a1, b2, acc[1][2]);
        acc[1][3] = wmma_bf16(a1, b3, acc[1][3]);
    }
}

// ---------------------------------------------------------------------------
// fp32 -> bf16 conversion (activations & weights), 8 elems/thread
// ---------------------------------------------------------------------------
__global__ __launch_bounds__(256)
void cvt_kernel(const float* __restrict__ src, bf16* __restrict__ dst, int n8) {
    const int i = blockIdx.x * 256 + threadIdx.x;
    if (i >= n8) return;
    v4f a = *(const v4f*)(src + (size_t)i * 8);
    v4f b = *(const v4f*)(src + (size_t)i * 8 + 4);
    v8bf o;
    o[0]=(bf16)a[0]; o[1]=(bf16)a[1]; o[2]=(bf16)a[2]; o[3]=(bf16)a[3];
    o[4]=(bf16)b[0]; o[5]=(bf16)b[1]; o[6]=(bf16)b[2]; o[7]=(bf16)b[3];
    *(v8bf*)(dst + (size_t)i * 8) = o;
}

// ---------------------------------------------------------------------------
// NT GEMM, all-bf16 operands, TDM-staged tiles, double-buffered on TENSORcnt.
// Out[m,n] = sum_k A[m,k]*W[n,k] + bias[n];  OUTF32 selects f32 vs bf16 store.
// ---------------------------------------------------------------------------
template<int OUTF32>
__global__ __launch_bounds__(256)
void gemm_nt_tdm(const bf16* __restrict__ A, const bf16* __restrict__ W,
                 const float* __restrict__ bias, void* __restrict__ Outp,
                 uint32_t Arows, uint32_t Wrows) {
    __shared__ __align__(16) bf16 As[2][128 * STR];
    __shared__ __align__(16) bf16 Bs[2][128 * STR];
    const int tid = threadIdx.x, lane = tid & 31, wave = tid >> 5;
    const int wm0 = (wave >> 1) * 32, wn0 = (wave & 1) * 64;
    const int m0 = blockIdx.y * 128, n0 = blockIdx.x * 128;
    const bf16* Abase = A + (size_t)m0 * D_MODEL;
    const bf16* Wbase = W + (size_t)n0 * D_MODEL;

    v8f acc[2][4] = {};

    if (wave == 0) {
        tdm_load_tile(Abase, lds_addr_of(As[0]), Arows);
        tdm_load_tile(Wbase, lds_addr_of(Bs[0]), Wrows);
    }
    const int nIter = D_MODEL / BK;   // 16
    for (int it = 0; it < nIter; ++it) {
        __syncthreads();              // next-buf readers from iter it-1 are done
        if (wave == 0) {
            if (it + 1 < nIter) {
                tdm_load_tile(Abase + (it + 1) * BK, lds_addr_of(As[(it + 1) & 1]), Arows);
                tdm_load_tile(Wbase + (it + 1) * BK, lds_addr_of(Bs[(it + 1) & 1]), Wrows);
                __builtin_amdgcn_s_wait_tensorcnt(2);   // current-buf pair retired
            } else {
                __builtin_amdgcn_s_wait_tensorcnt(0);
            }
        }
        __syncthreads();
        mma_tile(As[it & 1], Bs[it & 1], wm0, wn0, lane, acc);
    }
    // C layout: elem i -> row = i + (lane>=16)*8, col = lane&15
    const int colL = lane & 15, rsel = (lane >> 4) * 8;
    #pragma unroll
    for (int ms = 0; ms < 2; ++ms)
        #pragma unroll
        for (int ns = 0; ns < 4; ++ns) {
            const int gcol = n0 + wn0 + ns * 16 + colL;
            const float bv = bias[gcol];
            #pragma unroll
            for (int i = 0; i < 8; ++i) {
                const int grow = m0 + wm0 + ms * 16 + rsel + i;
                const float v = acc[ms][ns][i] + bv;
                if (OUTF32) ((float*)Outp)[(size_t)grow * D_MODEL + gcol] = v;
                else        ((bf16*)Outp)[(size_t)grow * D_MODEL + gcol] = (bf16)v;
            }
        }
}

// ---------------------------------------------------------------------------
// Circular correlation via Gram-matrix diagonal sums (TDM-staged tiles):
// corr[b,tau] += sum over tile elems M[t,s]=Q[t]·K[s] with (t-s) mod L == tau
// ---------------------------------------------------------------------------
__global__ __launch_bounds__(256)
void corr_kernel(const bf16* __restrict__ Q, const bf16* __restrict__ Kb,
                 float* __restrict__ corr) {
    __shared__ __align__(16) bf16 As[2][128 * STR];
    __shared__ __align__(16) bf16 Bs[2][128 * STR];
    __shared__ float diag[256];
    const int tid = threadIdx.x, lane = tid & 31, wave = tid >> 5;
    const int wm0 = (wave >> 1) * 32, wn0 = (wave & 1) * 64;
    const int b = blockIdx.z;
    const int t0 = blockIdx.y * 128, s0 = blockIdx.x * 128;
    const bf16* Abase = Q  + ((size_t)b * SEQ_L + t0) * D_MODEL;
    const bf16* Bbase = Kb + ((size_t)b * SEQ_L + s0) * D_MODEL;

    v8f acc[2][4] = {};

    if (wave == 0) {
        tdm_load_tile(Abase, lds_addr_of(As[0]), SEQ_L);
        tdm_load_tile(Bbase, lds_addr_of(Bs[0]), SEQ_L);
    }
    const int nIter = D_MODEL / BK;
    for (int it = 0; it < nIter; ++it) {
        __syncthreads();
        if (wave == 0) {
            if (it + 1 < nIter) {
                tdm_load_tile(Abase + (it + 1) * BK, lds_addr_of(As[(it + 1) & 1]), SEQ_L);
                tdm_load_tile(Bbase + (it + 1) * BK, lds_addr_of(Bs[(it + 1) & 1]), SEQ_L);
                __builtin_amdgcn_s_wait_tensorcnt(2);
            } else {
                __builtin_amdgcn_s_wait_tensorcnt(0);
            }
        }
        __syncthreads();
        mma_tile(As[it & 1], Bs[it & 1], wm0, wn0, lane, acc);
    }
    // reduce 128x128 tile into 255 circular diagonals (tau = t - s mod L)
    diag[tid] = 0.0f;
    __syncthreads();
    const int colL = lane & 15, rsel = (lane >> 4) * 8;
    #pragma unroll
    for (int ms = 0; ms < 2; ++ms)
        #pragma unroll
        for (int ns = 0; ns < 4; ++ns) {
            const int cl = wn0 + ns * 16 + colL;
            #pragma unroll
            for (int i = 0; i < 8; ++i) {
                const int rl = wm0 + ms * 16 + rsel + i;
                atomicAdd(&diag[(rl - cl) + 127], acc[ms][ns][i]);   // ds_add_f32
            }
        }
    __syncthreads();
    if (tid < 255) {
        const int tau = (t0 - s0 + tid - 127 + SEQ_L) & (SEQ_L - 1);
        atomicAdd(&corr[b * SEQ_L + tau], diag[tid]);                // global_atomic_add_f32
    }
}

// ---------------------------------------------------------------------------
// top-7 + softmax per batch (one block per batch)
// ---------------------------------------------------------------------------
__global__ __launch_bounds__(256)
void topk_kernel(const float* __restrict__ corr, float* __restrict__ wts,
                 int* __restrict__ dls) {
    __shared__ float sv[SEQ_L];
    __shared__ float rv[256];
    __shared__ int   ri[256];
    __shared__ float wloc[TOPK];
    __shared__ int   dloc[TOPK];
    const int b = blockIdx.x, tid = threadIdx.x;
    for (int i = tid; i < SEQ_L; i += 256)
        sv[i] = corr[b * SEQ_L + i] * (1.0f / 1024.0f);   // mean over H*E
    __syncthreads();
    for (int k = 0; k < TOPK; ++k) {
        float best = -1e30f; int bi = 0;
        for (int i = tid; i < SEQ_L; i += 256) {
            const float v = sv[i];
            if (v > best) { best = v; bi = i; }
        }
        rv[tid] = best; ri[tid] = bi;
        __syncthreads();
        for (int s = 128; s > 0; s >>= 1) {
            if (tid < s && rv[tid + s] > rv[tid]) { rv[tid] = rv[tid + s]; ri[tid] = ri[tid + s]; }
            __syncthreads();
        }
        if (tid == 0) { wloc[k] = rv[0]; dloc[k] = ri[0]; sv[ri[0]] = -1e30f; }
        __syncthreads();
    }
    if (tid == 0) {
        const float mx = wloc[0];
        float e[TOPK], s = 0.0f;
        for (int k = 0; k < TOPK; ++k) { e[k] = __expf(wloc[k] - mx); s += e[k]; }
        const float inv = 1.0f / s;
        for (int k = 0; k < TOPK; ++k) { wts[b * 8 + k] = e[k] * inv; dls[b * 8 + k] = dloc[k]; }
    }
}

// ---------------------------------------------------------------------------
// delay aggregation: Vagg[b,l,:] = sum_k w[b,k] * V[b,(l+delay[b,k])%L,:]
// ---------------------------------------------------------------------------
__global__ __launch_bounds__(128)
void agg_kernel(const bf16* __restrict__ V, const float* __restrict__ wts,
                const int* __restrict__ dls, bf16* __restrict__ Vagg) {
    __shared__ float w[TOPK];
    __shared__ int   dl[TOPK];
    const int r = blockIdx.x;              // 0..B*L-1
    const int b = r >> 11, l = r & (SEQ_L - 1);
    if (threadIdx.x < TOPK) { w[threadIdx.x] = wts[b * 8 + threadIdx.x];
                              dl[threadIdx.x] = dls[b * 8 + threadIdx.x]; }
    __syncthreads();
    const int d0 = threadIdx.x * 8;
    float acc[8] = {};
    #pragma unroll
    for (int k = 0; k < TOPK; ++k) {
        const int src = (l + dl[k]) & (SEQ_L - 1);
        v8bf v = *(const v8bf*)(V + ((size_t)b * SEQ_L + src) * D_MODEL + d0);
        const float wk = w[k];
        #pragma unroll
        for (int i = 0; i < 8; ++i) acc[i] += wk * (float)v[i];
    }
    v8bf o;
    #pragma unroll
    for (int i = 0; i < 8; ++i) o[i] = (bf16)acc[i];
    *(v8bf*)(Vagg + (size_t)r * D_MODEL + d0) = o;
}

// ---------------------------------------------------------------------------
extern "C" void kernel_launch(void* const* d_in, const int* in_sizes, int n_in,
                              void* d_out, int out_size, void* d_ws, size_t ws_size,
                              hipStream_t stream) {
    const float* q_in = (const float*)d_in[0];
    const float* k_in = (const float*)d_in[1];
    const float* v_in = (const float*)d_in[2];
    const float* Wq   = (const float*)d_in[3];
    const float* bq   = (const float*)d_in[4];
    const float* Wk   = (const float*)d_in[5];
    const float* bk   = (const float*)d_in[6];
    const float* Wv   = (const float*)d_in[7];
    const float* bv   = (const float*)d_in[8];
    const float* Wo   = (const float*)d_in[9];
    const float* bo   = (const float*)d_in[10];

    const size_t QSZ = (size_t)BATCH * SEQ_L * D_MODEL * sizeof(bf16);  // 32 MB
    const size_t WSZ = (size_t)D_MODEL * D_MODEL * sizeof(bf16);        // 2 MB
    char* ws = (char*)d_ws;
    bf16*  Qbf  = (bf16*)(ws);
    bf16*  Kbf  = (bf16*)(ws + QSZ);
    bf16*  Vbf  = (bf16*)(ws + 2 * QSZ);
    bf16*  ABUF = (bf16*)(ws + 3 * QSZ);           // staged activations / Vagg
    bf16*  WBUF = (bf16*)(ws + 4 * QSZ);           // staged weights
    float* corr = (float*)(ws + 4 * QSZ + WSZ);
    float* wts  = corr + BATCH * SEQ_L;
    int*   dls  = (int*)(wts + BATCH * 8);

    const int ACT8 = (BATCH * SEQ_L * D_MODEL) / 8;    // 2,097,152
    const int WT8  = (D_MODEL * D_MODEL) / 8;          // 131,072
    const dim3 gGemm(D_MODEL / 128, (BATCH * SEQ_L) / 128);  // (8, 128)

    hipMemsetAsync(corr, 0, (size_t)BATCH * SEQ_L * sizeof(float), stream);

    // Q projection
    cvt_kernel<<<ACT8 / 256, 256, 0, stream>>>(q_in, ABUF, ACT8);
    cvt_kernel<<<WT8 / 256, 256, 0, stream>>>(Wq, WBUF, WT8);
    gemm_nt_tdm<0><<<gGemm, 256, 0, stream>>>(ABUF, WBUF, bq, Qbf, BATCH * SEQ_L, D_MODEL);
    // K projection
    cvt_kernel<<<ACT8 / 256, 256, 0, stream>>>(k_in, ABUF, ACT8);
    cvt_kernel<<<WT8 / 256, 256, 0, stream>>>(Wk, WBUF, WT8);
    gemm_nt_tdm<0><<<gGemm, 256, 0, stream>>>(ABUF, WBUF, bk, Kbf, BATCH * SEQ_L, D_MODEL);
    // V projection
    cvt_kernel<<<ACT8 / 256, 256, 0, stream>>>(v_in, ABUF, ACT8);
    cvt_kernel<<<WT8 / 256, 256, 0, stream>>>(Wv, WBUF, WT8);
    gemm_nt_tdm<0><<<gGemm, 256, 0, stream>>>(ABUF, WBUF, bv, Vbf, BATCH * SEQ_L, D_MODEL);

    // autocorrelation
    corr_kernel<<<dim3(SEQ_L / 128, SEQ_L / 128, BATCH), 256, 0, stream>>>(Qbf, Kbf, corr);
    topk_kernel<<<BATCH, 256, 0, stream>>>(corr, wts, dls);
    agg_kernel<<<BATCH * SEQ_L, 128, 0, stream>>>(Vbf, wts, dls, ABUF);  // ABUF = Vagg

    // output projection (f32 out)
    cvt_kernel<<<WT8 / 256, 256, 0, stream>>>(Wo, WBUF, WT8);
    gemm_nt_tdm<1><<<gGemm, 256, 0, stream>>>(ABUF, WBUF, bo, (float*)d_out, BATCH * SEQ_L, D_MODEL);
}
